// POSKahler_51745765982234
// MI455X (gfx1250) — matrix operand order repaired and checked by reading, not compiled, not run on hardware
//
#include <hip/hip_runtime.h>
#include <math.h>

#define K_MON   165
#define KP      176      // padded half (x | y)
#define TWOKP   352      // GEMM K depth = 2*KP = 11 chunks of 32
#define USTRIDE 360      // padded f16 stride (16B aligned, LDS-conflict-free)
#define NTILES  22       // 352 / 16 output column tiles
#define KCHUNKS 11       // 352 / 32
#define ROWS_PER_BLOCK 64
#define THREADS 128

typedef __attribute__((ext_vector_type(16))) _Float16 v16h;
typedef __attribute__((ext_vector_type(8)))  _Float16 v8h;
typedef __attribute__((ext_vector_type(8)))  float    v8f;

// ---- monomial power table: product(range(4), repeat=8) filtered sum<=3, lex order ----
struct Pows { unsigned short v[K_MON]; };
constexpr Pows makePows() {
  Pows t{};
  int k = 0;
  for (int a0 = 0; a0 <= 3; a0++)
  for (int a1 = 0; a1 <= 3 - a0; a1++)
  for (int a2 = 0; a2 <= 3 - a0 - a1; a2++)
  for (int a3 = 0; a3 <= 3 - a0 - a1 - a2; a3++)
  for (int a4 = 0; a4 <= 3 - a0 - a1 - a2 - a3; a4++)
  for (int a5 = 0; a5 <= 3 - a0 - a1 - a2 - a3 - a4; a5++)
  for (int a6 = 0; a6 <= 3 - a0 - a1 - a2 - a3 - a4 - a5; a6++)
  for (int a7 = 0; a7 <= 3 - a0 - a1 - a2 - a3 - a4 - a5 - a6; a7++) {
    t.v[k++] = (unsigned short)(a0 | (a1 << 2) | (a2 << 4) | (a3 << 6) |
                                (a4 << 8) | (a5 << 10) | (a6 << 12) | (a7 << 14));
  }
  return t;
}
__constant__ const Pows POWPACK = makePows();

// ---- build M = [[Lr, -Li],[Li, Lr]] (column-major, f16, padded 352 cols x 360 rows) ----
__global__ void build_M_kernel(const float* __restrict__ Lre,
                               const float* __restrict__ Lim,
                               _Float16* __restrict__ M) {
  int idx = blockIdx.x * blockDim.x + threadIdx.x;
  const int total = TWOKP * USTRIDE;
  if (idx >= total) return;
  int n = idx / USTRIDE;   // column 0..351 (output index of w = M^T u)
  int k = idx % USTRIDE;   // row    0..359 (input index into u)
  int c = n / KP, m = n % KP;
  int r = k / KP, j = k % KP;
  float val = 0.f;
  if (k < TWOKP && m < K_MON && j < K_MON) {
    float lr = 0.f, li = 0.f;
    if (j > m)      { lr = Lre[j * K_MON + m]; li = Lim[j * K_MON + m]; }
    else if (j == m) {
      float d  = Lre[j * K_MON + j];
      float sp = (d > 20.f) ? d : log1pf(expf(d));   // softplus
      lr = sp + 0.001f;
    }
    // blocks: (0,0)=Lr (0,1)=-Li (1,0)=Li (1,1)=Lr
    val = (r == c) ? lr : ((r == 0) ? -li : li);
  }
  M[n * USTRIDE + k] = (_Float16)val;
}

// async wait helper
__device__ __forceinline__ void async_wait0() {
#if __has_builtin(__builtin_amdgcn_s_wait_asynccnt)
  __builtin_amdgcn_s_wait_asynccnt(0);
#else
  asm volatile("s_wait_asynccnt 0" ::: "memory");
#endif
}

// issue async global->LDS copy of one 16-column M slab (11520 B = 720 x 16B)
__device__ __forceinline__ void stage_async(const _Float16* __restrict__ Mg, int nt,
                                            _Float16* buf, int t) {
  const _Float16* src = Mg + (size_t)nt * 16 * USTRIDE;
#pragma unroll
  for (int i = 0; i < 6; i++) {
    int idx = t + i * THREADS;
    if (idx < 720) {
      unsigned la = (unsigned)(unsigned long long)(buf + idx * 8);
      unsigned long long ga = (unsigned long long)(src + idx * 8);
      asm volatile("global_load_async_to_lds_b128 %0, %1, off"
                   :: "v"(la), "v"(ga) : "memory");
    }
  }
}

// ---- main: monomials -> LDS -> Areg, W = U*M via WMMA, quad = rowsum(W^2), log ----
__global__ void __launch_bounds__(THREADS)
poskahler_main(const float* __restrict__ q, const _Float16* __restrict__ Mg,
               float* __restrict__ out) {
  // U (46080 B) is used once to shuffle monomials into A-fragments, then the
  // same LDS is reused as the double-buffered B stage (2 x 11520 B), separated
  // by barriers (which flush DScnt/ASYNCcnt on this target).
  __shared__ __align__(16) _Float16 SH[ROWS_PER_BLOCK * USTRIDE];
  _Float16* U = SH;
  _Float16* Bsh0 = SH;
  _Float16* Bsh1 = SH + 16 * USTRIDE;

  const int t    = threadIdx.x;
  const int lane = t & 31;
  const int wave = t >> 5;

  // ---------- Phase 1: monomials (2 threads per sample) ----------
  {
    int s = t >> 1;        // block-local sample 0..63
    int h = t & 1;         // which half of the k-range
    long row = (long)blockIdx.x * ROWS_PER_BLOCK + s;
    const float* qr = q + row * 16;
    float x[8], y[8], l2r[8], l2i[8], l3r[8], l3i[8];
#pragma unroll
    for (int c = 0; c < 8; c++) { x[c] = qr[c]; y[c] = qr[c + 8]; }
#pragma unroll
    for (int c = 0; c < 8; c++) {
      l2r[c] = x[c] * x[c] - y[c] * y[c];
      l2i[c] = 2.f * x[c] * y[c];
      l3r[c] = l2r[c] * x[c] - l2i[c] * y[c];
      l3i[c] = l2r[c] * y[c] + l2i[c] * x[c];
    }
    int kb = h ? 83 : 0;
    int ke = h ? K_MON : 83;
    for (int k = kb; k < ke; k++) {
      unsigned pw = POWPACK.v[k];
      float mr = 1.f, mi = 0.f;
#pragma unroll
      for (int c = 0; c < 8; c++) {
        int p = (pw >> (2 * c)) & 3;
        if (p) {
          float zr = (p == 1) ? x[c] : ((p == 2) ? l2r[c] : l3r[c]);
          float zi = (p == 1) ? y[c] : ((p == 2) ? l2i[c] : l3i[c]);
          float nr = mr * zr - mi * zi;
          float ni = mr * zi + mi * zr;
          mr = nr; mi = ni;
        }
      }
      U[s * USTRIDE + k]      = (_Float16)mr;   // x-part
      U[s * USTRIDE + KP + k] = (_Float16)mi;   // y-part
    }
    // zero the 11-element pads (h==0 -> [165,176), h==1 -> [341,352))
    int pbase = s * USTRIDE + h * KP + K_MON;
    for (int k = 0; k < KP - K_MON; k++) U[pbase + k] = (_Float16)0.f;
  }

  __syncthreads();

  // ---------- Phase 2a: persist A-fragments in registers ----------
  const int s16  = lane & 15;
  const int half = lane >> 4;
  const int aRow = (wave * 16 + s16) * USTRIDE;

  v16h Areg[KCHUNKS];
#pragma unroll
  for (int kc = 0; kc < KCHUNKS; kc++) {
    // lane<16 -> k0+{0..7,16..23}; lane>=16 -> k0+{8..15,24..31}
    const v8h* ap = (const v8h*)&U[aRow + kc * 32 + 8 * half];
    v8h a0 = ap[0];
    v8h a1 = ap[2];               // +16 f16
#pragma unroll
    for (int i = 0; i < 8; i++) { Areg[kc][i] = a0[i]; Areg[kc][8 + i] = a1[i]; }
  }

  __syncthreads();   // all waves done reading U before async writes reuse the LDS

  // ---------- Phase 2b: WMMA GEMM with double-buffered async B staging ----------
  float psum[8];
#pragma unroll
  for (int i = 0; i < 8; i++) psum[i] = 0.f;

  stage_async(Mg, 0, Bsh0, t);
  async_wait0();
  __syncthreads();

  int buf = 0;
  for (int nt = 0; nt < NTILES; nt++) {
    if (nt + 1 < NTILES)
      stage_async(Mg, nt + 1, (buf ? Bsh0 : Bsh1), t);   // overlap with compute

    const _Float16* bc = buf ? Bsh1 : Bsh0;
    v8f acc = {};
#pragma unroll
    for (int kc = 0; kc < KCHUNKS; kc++) {
      // B-frag: column s16, K rows k0+8*half+{0..7,16..23} (matches A table order)
      const v8h* bp = (const v8h*)&bc[s16 * USTRIDE + kc * 32 + 8 * half];
      v8h b0 = bp[0];
      v8h b1 = bp[2];
      v16h B;
#pragma unroll
      for (int i = 0; i < 8; i++) { B[i] = b0[i]; B[8 + i] = b1[i]; }
      acc = __builtin_amdgcn_wmma_f32_16x16x32_f16(false, Areg[kc], false, B,
                                                   (short)0, acc, false, false);
    }
#pragma unroll
    for (int i = 0; i < 8; i++) psum[i] += acc[i] * acc[i];

    if (nt + 1 < NTILES) {
      async_wait0();
      __syncthreads();
      buf ^= 1;
    }
  }

  // butterfly reduction over each 16-lane group (columns of the D tile)
#pragma unroll
  for (int m = 1; m < 16; m <<= 1) {
#pragma unroll
    for (int i = 0; i < 8; i++) psum[i] += __shfl_xor(psum[i], m, 32);
  }
  if ((lane & 15) == 0) {
    long rowBase = (long)blockIdx.x * ROWS_PER_BLOCK + wave * 16 + half * 8;
#pragma unroll
    for (int i = 0; i < 8; i++) out[rowBase + i] = logf(psum[i] + 1e-12f);
  }
}

extern "C" void kernel_launch(void* const* d_in, const int* in_sizes, int n_in,
                              void* d_out, int out_size, void* d_ws, size_t ws_size,
                              hipStream_t stream) {
  const float* q   = (const float*)d_in[0];
  const float* Lre = (const float*)d_in[1];
  const float* Lim = (const float*)d_in[2];
  _Float16* M = (_Float16*)d_ws;              // needs 352*360*2 = 253440 B
  (void)n_in; (void)ws_size; (void)out_size;

  {
    const int total = TWOKP * USTRIDE;
    build_M_kernel<<<(total + 255) / 256, 256, 0, stream>>>(Lre, Lim, M);
  }
  int B = in_sizes[0] / 16;                   // 262144
  int blocks = B / ROWS_PER_BLOCK;            // 4096
  poskahler_main<<<blocks, THREADS, 0, stream>>>(q, M, (float*)d_out);
}
